// MHSAModule_25151328485865
// MI455X (gfx1250) — matrix operand order repaired and checked by reading, not compiled, not run on hardware
//
#include <hip/hip_runtime.h>

typedef __attribute__((ext_vector_type(16))) __bf16        v16bf;
typedef __attribute__((ext_vector_type(8)))  float         v8f;
typedef __attribute__((ext_vector_type(4)))  unsigned int  u32x4;
typedef __attribute__((ext_vector_type(8)))  unsigned int  u32x8;

#define B_ 8
#define T_ 1024
#define D_ 512
#define H_ 8
#define K_ 64
#define L_ 2047

#if __has_builtin(__builtin_amdgcn_wave_barrier)
#define WAVE_BAR() __builtin_amdgcn_wave_barrier()
#else
#define WAVE_BAR()
#endif

#if __has_builtin(__builtin_amdgcn_s_wait_tensorcnt)
#define WAIT_TENSOR0() __builtin_amdgcn_s_wait_tensorcnt(0)
#else
#define WAIT_TENSOR0() asm volatile("s_wait_tensorcnt 0x0" ::: "memory")
#endif

__device__ __forceinline__ unsigned short f2bf(float f) {
    unsigned int u = __float_as_uint(f);
    unsigned int r = (u + 0x7fffu + ((u >> 16) & 1u)) >> 16;
    return (unsigned short)r;
}

union ABf {
    v16bf        v;
    u32x4        u[2];
    unsigned int w[8];
};

__device__ __forceinline__ v8f wmma_bf16(v16bf a, v16bf b, v8f c) {
    return __builtin_amdgcn_wmma_f32_16x16x32_bf16(false, a, false, b, (short)0, c, false, false);
}

// Issue a 2D Tensor-Data-Mover load described by SGPR groups g0 (4 sgpr) / g1 (8 sgpr).
__device__ __forceinline__ void tdm_load(u32x4 g0, u32x8 g1) {
    asm volatile("tensor_load_to_lds %0, %1" :: "s"(g0), "s"(g1) : "memory");
}

// ---------------------------------------------------------------------------
// LayerNorm (fp32 in) -> bf16 out.  One 256-thread block (8 wave32) per row.
// ---------------------------------------------------------------------------
__global__ __launch_bounds__(256) void ln_kernel(const float* __restrict__ x,
                                                 const float* __restrict__ gamma,
                                                 const float* __restrict__ beta,
                                                 unsigned short* __restrict__ y) {
    const int row = blockIdx.x;
    const int tid = threadIdx.x;
    const float a = x[(size_t)row * D_ + tid];
    const float b = x[(size_t)row * D_ + 256 + tid];
    float s = a + b;
    float q = a * a + b * b;
#pragma unroll
    for (int off = 16; off >= 1; off >>= 1) {
        s += __shfl_xor(s, off, 32);
        q += __shfl_xor(q, off, 32);
    }
    __shared__ float rs[8], rq[8];
    const int wid = tid >> 5, lane = tid & 31;
    if (lane == 0) { rs[wid] = s; rq[wid] = q; }
    __syncthreads();
    float S = 0.f, Q = 0.f;
#pragma unroll
    for (int i = 0; i < 8; ++i) { S += rs[i]; Q += rq[i]; }
    const float mu   = S * (1.0f / D_);
    const float var  = Q * (1.0f / D_) - mu * mu;
    const float rstd = rsqrtf(var + 1e-3f);
    y[(size_t)row * D_ + tid]       = f2bf((a - mu) * rstd * gamma[tid]       + beta[tid]);
    y[(size_t)row * D_ + 256 + tid] = f2bf((b - mu) * rstd * gamma[tid + 256] + beta[tid + 256]);
}

// ---------------------------------------------------------------------------
// Elementwise fp32 -> bf16 cast (rel_pos).
// ---------------------------------------------------------------------------
__global__ __launch_bounds__(256) void cast_kernel(const float* __restrict__ in,
                                                   unsigned short* __restrict__ out, int n) {
    int i = blockIdx.x * 256 + threadIdx.x;
    if (i < n) out[i] = f2bf(in[i]);
}

// ---------------------------------------------------------------------------
// Transpose + cast: W[R,C] fp32 -> Wt[C,R] bf16 (LDS tiled, R=C=512 here).
// ---------------------------------------------------------------------------
__global__ __launch_bounds__(256) void transpose_cast(const float* __restrict__ W,
                                                      unsigned short* __restrict__ Wt,
                                                      int R, int C) {
    __shared__ float tile[32][33];
    const int tx = threadIdx.x, ty = threadIdx.y;
    const int bx = blockIdx.x * 32, by = blockIdx.y * 32;
#pragma unroll
    for (int i = 0; i < 4; ++i)
        tile[ty + 8 * i][tx] = W[(size_t)(by + ty + 8 * i) * C + bx + tx];
    __syncthreads();
#pragma unroll
    for (int i = 0; i < 4; ++i)
        Wt[(size_t)(bx + ty + 8 * i) * R + by + tx] = f2bf(tile[tx][ty + 8 * i]);
}

// ---------------------------------------------------------------------------
// bf16 transpose of v: [B*T, H*K] -> vT[B,H,K,T], staged via async global->LDS
// (ASYNCcnt path): each of the first 128 lanes issues one global_load_async_to_lds_b128.
// Flat 256-thread block; tile 32 (s) x 32 (k), LDS row stride 40 ushorts.
// ---------------------------------------------------------------------------
__global__ __launch_bounds__(256) void transpose_v(const unsigned short* __restrict__ v,
                                                   unsigned short* __restrict__ vT) {
    __shared__ __align__(16) unsigned short tile[32 * 40];
    const int tid = threadIdx.x;
    const int bi = blockIdx.z / H_, h = blockIdx.z % H_;
    const int s0 = blockIdx.x * 32, k0 = blockIdx.y * 32;
    const unsigned lbase = (unsigned)(size_t)(void*)tile;

    if (tid < 128) {
        const int r = tid >> 2, c = tid & 3;  // 16B per lane, 4 lanes per row
        const unsigned short* g =
            v + ((size_t)(bi * T_ + s0 + r)) * D_ + h * K_ + k0 + c * 8;
        const unsigned ldst = lbase + (unsigned)(r * 80 + c * 16);
        asm volatile("global_load_async_to_lds_b128 %0, %1, off"
                     :: "v"(ldst), "v"((unsigned long long)(size_t)g) : "memory");
    }
    asm volatile("s_wait_asynccnt 0x0" ::: "memory");
    __syncthreads();

    const int tx = tid & 31, ty = tid >> 5;
#pragma unroll
    for (int i = 0; i < 4; ++i) {
        const int kk = ty + 8 * i;
        vT[((size_t)((bi * H_ + h) * K_) + k0 + kk) * T_ + s0 + tx] = tile[tx * 40 + kk];
    }
}

// ---------------------------------------------------------------------------
// Generic bf16 WMMA GEMM: C[M,N] = A[M,Kd] * Bt[N,Kd]^T, fp32 accumulate.
// Epilogues: (out0[,bias0])(out1[,bias1]) bf16, or outf fp32 (+resid).
// Block = 256 threads = 8 waves (4 M-tiles x 2 N-tiles of 16x16).
// ---------------------------------------------------------------------------
__global__ __launch_bounds__(256) void gemm_bf16_wmma(
    const unsigned short* __restrict__ A, const unsigned short* __restrict__ Bt,
    int M, int N, int Kd,
    unsigned short* __restrict__ out0, const float* __restrict__ bias0,
    unsigned short* __restrict__ out1, const float* __restrict__ bias1,
    float* __restrict__ outf, const float* __restrict__ resid) {
    const int wid  = threadIdx.x >> 5;
    const int lane = threadIdx.x & 31;
    const int hi   = lane >> 4;
    const int lm   = lane & 15;
    const int wm = wid & 3, wn = wid >> 2;
    const int m0 = blockIdx.x * 64 + wm * 16;
    const int n0 = blockIdx.y * 32 + wn * 16;
    int arow = m0 + lm; if (arow >= M) arow = M - 1;
    const int ncol = n0 + lm;

    v8f acc = {0.f, 0.f, 0.f, 0.f, 0.f, 0.f, 0.f, 0.f};
    for (int kk = 0; kk < Kd; kk += 32) {
        ABf a, b;
        const u32x4* ap = (const u32x4*)(A + (size_t)arow * Kd + kk + (hi ? 16 : 0));
        a.u[0] = ap[0];
        a.u[1] = ap[1];  // +8 elements
        const u32x4* bp = (const u32x4*)(Bt + (size_t)ncol * Kd + kk + (hi ? 16 : 0));
        b.u[0] = bp[0];
        b.u[1] = bp[1];
        __builtin_prefetch(ap + 4, 0, 1);
        __builtin_prefetch(bp + 4, 0, 1);
        acc = wmma_bf16(a.v, b.v, acc);
    }
#pragma unroll
    for (int r = 0; r < 8; ++r) {
        const int row = m0 + r + 8 * hi;
        if (row < M) {
            const float v = acc[r];
            const size_t idx = (size_t)row * N + ncol;
            if (outf) {
                outf[idx] = v + (resid ? resid[idx] : 0.f);
            } else {
                out0[idx] = f2bf(v + (bias0 ? bias0[ncol] : 0.f));
                if (out1) out1[idx] = f2bf(v + (bias1 ? bias1[ncol] : 0.f));
            }
        }
    }
}

// ---------------------------------------------------------------------------
// Fused rel-pos attention, flash-style.  Grid (T/64, H, B), block = 4 wave32.
// Each wave owns 16 queries; loops over key tiles of 32.
// K/V tiles are DMA'd once per block into LDS by the Tensor Data Mover
// (TENSORcnt path) and consumed by all 4 waves via ds_load_b128.
// bd term: dense 16x48 band GEMM + LDS shifted read (Transformer-XL shift).
// ---------------------------------------------------------------------------
__global__ __launch_bounds__(128) void attn_kernel(
    const unsigned short* __restrict__ qu, const unsigned short* __restrict__ qv,
    const unsigned short* __restrict__ kb, const unsigned short* __restrict__ pb,
    const unsigned short* __restrict__ vT, unsigned short* __restrict__ o) {
    __shared__ float smem[4][16 * 48];
    __shared__ __align__(16) unsigned short ktile[32 * 64];  // [key][feat]
    __shared__ __align__(16) unsigned short vtile[64 * 32];  // [feat][key]

    const int w    = threadIdx.x >> 5;
    const int lane = threadIdx.x & 31;
    const int hi   = lane >> 4;
    const int lm   = lane & 15;
    const int bi = blockIdx.z, h = blockIdx.y;
    const int t0 = blockIdx.x * 64 + w * 16;
    const float scale = 0.125f;  // 1/sqrt(64)

    // Loop-invariant TDM descriptor group-1 (tensor geometry), 2-byte elements.
    // k view: [B*T rows, 512] row-major;  vT view: [B*H*64 rows, T] row-major.
    const u32x8 g1k = {0x00010000u, (unsigned)D_ << 16, (unsigned)(B_ * T_) << 16,
                       64u << 16, 32u, (unsigned)D_, 0u, 0u};
    const u32x8 g1v = {0x00010000u, (unsigned)T_ << 16, (unsigned)(B_ * H_ * K_) << 16,
                       32u << 16, 64u, (unsigned)T_, 0u, 0u};
    const unsigned lds_k = (unsigned)(size_t)(void*)ktile;
    const unsigned lds_v = (unsigned)(size_t)(void*)vtile;

    // Load per-wave A operands (q+u, q+v): 16 queries x 64 features = 2 tiles each.
    ABf quA[2], qvA[2];
    {
        const int tq = t0 + lm;
#pragma unroll
        for (int c = 0; c < 2; ++c) {
            const u32x4* ap = (const u32x4*)(qu + ((size_t)(bi * T_ + tq)) * D_ + h * K_ + 32 * c + (hi ? 16 : 0));
            quA[c].u[0] = ap[0]; quA[c].u[1] = ap[1];
            const u32x4* vp = (const u32x4*)(qv + ((size_t)(bi * T_ + tq)) * D_ + h * K_ + 32 * c + (hi ? 16 : 0));
            qvA[c].u[0] = vp[0]; qvA[c].u[1] = vp[1];
        }
    }

    v8f O[4];
    const v8f vzero = {0.f, 0.f, 0.f, 0.f, 0.f, 0.f, 0.f, 0.f};
#pragma unroll
    for (int nt = 0; nt < 4; ++nt) O[nt] = vzero;
    float rmax[8], rsum[8];
#pragma unroll
    for (int r = 0; r < 8; ++r) { rmax[r] = -3.0e38f; rsum[r] = 0.f; }

    for (int s0 = 0; s0 < T_; s0 += 32) {
        // ---- TDM: stage k tile [32 keys x 64 feats] and v tile [64 x 32] ----
        if (w == 0) {
            const unsigned long long ak =
                (unsigned long long)(size_t)(kb + ((size_t)(bi * T_ + s0)) * D_ + h * K_);
            const unsigned long long av =
                (unsigned long long)(size_t)(vT + ((size_t)(bi * H_ + h) * K_) * T_ + s0);
            const u32x4 g0k = {1u, lds_k, (unsigned)ak,
                               ((unsigned)((ak >> 32) & 0x01FFFFFFu)) | 0x80000000u};
            const u32x4 g0v = {1u, lds_v, (unsigned)av,
                               ((unsigned)((av >> 32) & 0x01FFFFFFu)) | 0x80000000u};
            tdm_load(g0k, g1k);
            tdm_load(g0v, g1v);
            WAIT_TENSOR0();
        }
        __syncthreads();

        // ---- content term ac: two 16x16 score tiles over K=64 ----
        v8f S[2];
        S[0] = vzero; S[1] = vzero;
#pragma unroll
        for (int st = 0; st < 2; ++st) {
#pragma unroll
            for (int c = 0; c < 2; ++c) {
                ABf b;
                const u32x4* kp = (const u32x4*)((const char*)ktile +
                                                 (16 * st + lm) * 128 + 64 * c + 32 * hi);
                b.u[0] = kp[0]; b.u[1] = kp[1];
                S[st] = wmma_bf16(quA[c].v, b.v, S[st]);
            }
        }
        // ---- positional band: 16x48 GEMM vs p rows l0..l0+47 ----
        const int l0 = s0 - t0 + T_ - 16;
#pragma unroll
        for (int jt = 0; jt < 3; ++jt) {
            v8f Bd = vzero;
            int l = l0 + 16 * jt + lm;
            if (l < 0) l = 0;
            if (l > L_ - 1) l = L_ - 1;
#pragma unroll
            for (int c = 0; c < 2; ++c) {
                ABf b;
                const u32x4* bp = (const u32x4*)(pb + (size_t)l * D_ + h * K_ + 32 * c + (hi ? 16 : 0));
                b.u[0] = bp[0]; b.u[1] = bp[1];
                Bd = wmma_bf16(qvA[c].v, b.v, Bd);
            }
#pragma unroll
            for (int r = 0; r < 8; ++r)
                smem[w][(r + 8 * hi) * 48 + 16 * jt + lm] = Bd[r];
        }
        WAVE_BAR();
        // ---- rel-shift read + scale ----
#pragma unroll
        for (int st = 0; st < 2; ++st)
#pragma unroll
            for (int r = 0; r < 8; ++r) {
                const int row = r + 8 * hi;
                const int j   = (16 * st + lm) + 15 - row;
                S[st][r] = (S[st][r] + smem[w][row * 48 + j]) * scale;
            }
        WAVE_BAR();
        // ---- online softmax ----
#pragma unroll
        for (int r = 0; r < 8; ++r) {
            float mloc = fmaxf(S[0][r], S[1][r]);
#pragma unroll
            for (int off = 8; off >= 1; off >>= 1) mloc = fmaxf(mloc, __shfl_xor(mloc, off, 16));
            const float mnew  = fmaxf(rmax[r], mloc);
            const float alpha = __expf(rmax[r] - mnew);
            rmax[r] = mnew;
            S[0][r] = __expf(S[0][r] - mnew);
            S[1][r] = __expf(S[1][r] - mnew);
            float ps = S[0][r] + S[1][r];
#pragma unroll
            for (int off = 8; off >= 1; off >>= 1) ps += __shfl_xor(ps, off, 16);
            rsum[r] = rsum[r] * alpha + ps;
#pragma unroll
            for (int nt = 0; nt < 4; ++nt) O[nt][r] *= alpha;
        }
        // ---- stage P (bf16) and repack to WMMA A layout via LDS ----
        unsigned short* psm = (unsigned short*)&smem[w][0];
#pragma unroll
        for (int r = 0; r < 8; ++r) {
            const int row = r + 8 * hi;
            psm[row * 32 + lm]      = f2bf(S[0][r]);
            psm[row * 32 + 16 + lm] = f2bf(S[1][r]);
        }
        WAVE_BAR();
        ABf pa;
        {
            const unsigned int* psu = (const unsigned int*)&smem[w][0];
            const int base = lm * 16 + (hi ? 8 : 0);
#pragma unroll
            for (int vv = 0; vv < 4; ++vv) {
                pa.w[vv]     = psu[base + vv];
                pa.w[4 + vv] = psu[base + 4 + vv];
            }
        }
        WAVE_BAR();
        // ---- O += P * V  (K = 32 keys, N = 64 features) ----
#pragma unroll
        for (int nt = 0; nt < 4; ++nt) {
            ABf b;
            const u32x4* vp = (const u32x4*)((const char*)vtile + (16 * nt + lm) * 64 + 32 * hi);
            b.u[0] = vp[0]; b.u[1] = vp[1];
            O[nt] = wmma_bf16(pa.v, b.v, O[nt]);
        }
        __syncthreads();  // all waves done with ktile/vtile before next TDM
    }

    // normalize and store o (bf16, layout [B*T, H*K])
#pragma unroll
    for (int nt = 0; nt < 4; ++nt)
#pragma unroll
        for (int r = 0; r < 8; ++r) {
            const int t = t0 + r + 8 * hi;
            const float val = O[nt][r] * (1.0f / rsum[r]);
            o[((size_t)(bi * T_ + t)) * D_ + h * K_ + 16 * nt + lm] = f2bf(val);
        }
}

// ---------------------------------------------------------------------------
extern "C" void kernel_launch(void* const* d_in, const int* in_sizes, int n_in,
                              void* d_out, int out_size, void* d_ws, size_t ws_size,
                              hipStream_t stream) {
    (void)in_sizes; (void)n_in; (void)out_size; (void)ws_size;
    const float* x     = (const float*)d_in[0];
    const float* rel   = (const float*)d_in[1];
    const float* ubias = (const float*)d_in[2];
    const float* vbias = (const float*)d_in[3];
    const float* Wq    = (const float*)d_in[4];
    const float* Wk    = (const float*)d_in[5];
    const float* Wv    = (const float*)d_in[6];
    const float* Wp    = (const float*)d_in[7];
    const float* Wo    = (const float*)d_in[8];
    const float* gamma = (const float*)d_in[9];
    const float* beta  = (const float*)d_in[10];
    float* out = (float*)d_out;

    char* ws = (char*)d_ws;
    const size_t SZ_BT  = (size_t)B_ * T_ * D_ * 2;   // 8 MiB bf16 [B*T, 512]
    const size_t SZ_W   = (size_t)D_ * D_ * 2;        // 512 KiB bf16
    const size_t SZ_REL = 2097152;                    // >= L*D*2, padded

    size_t off = 0;
    unsigned short* ybf  = (unsigned short*)(ws + off); off += SZ_BT;
    unsigned short* Wqt  = (unsigned short*)(ws + off); off += SZ_W;
    unsigned short* Wkt  = (unsigned short*)(ws + off); off += SZ_W;
    unsigned short* Wvt  = (unsigned short*)(ws + off); off += SZ_W;
    unsigned short* Wpt  = (unsigned short*)(ws + off); off += SZ_W;
    unsigned short* Wot  = (unsigned short*)(ws + off); off += SZ_W;
    unsigned short* rlbf = (unsigned short*)(ws + off); off += SZ_REL;
    unsigned short* qubf = (unsigned short*)(ws + off); off += SZ_BT;
    unsigned short* qvbf = (unsigned short*)(ws + off); off += SZ_BT;
    unsigned short* kbf  = (unsigned short*)(ws + off); off += SZ_BT;
    unsigned short* vbf  = (unsigned short*)(ws + off); off += SZ_BT;
    unsigned short* vTb  = (unsigned short*)(ws + off); off += SZ_BT;
    unsigned short* pbf  = (unsigned short*)(ws + off); off += SZ_REL;
    unsigned short* obf  = (unsigned short*)(ws + off); off += SZ_BT;

    // 1) LayerNorm -> bf16
    ln_kernel<<<B_ * T_, 256, 0, stream>>>(x, gamma, beta, ybf);
    // 2) rel_pos cast
    cast_kernel<<<(L_ * D_ + 255) / 256, 256, 0, stream>>>(rel, rlbf, L_ * D_);
    // 3) weight transpose+cast (all 512x512)
    dim3 tg(16, 16), tb(32, 8);
    transpose_cast<<<tg, tb, 0, stream>>>(Wq, Wqt, D_, D_);
    transpose_cast<<<tg, tb, 0, stream>>>(Wk, Wkt, D_, D_);
    transpose_cast<<<tg, tb, 0, stream>>>(Wv, Wvt, D_, D_);
    transpose_cast<<<tg, tb, 0, stream>>>(Wp, Wpt, D_, D_);
    transpose_cast<<<tg, tb, 0, stream>>>(Wo, Wot, D_, D_);
    // 4) projections (WMMA GEMMs)
    dim3 gq((B_ * T_) / 64, D_ / 32);
    gemm_bf16_wmma<<<gq, 256, 0, stream>>>(ybf, Wqt, B_ * T_, D_, D_,
                                           qubf, ubias, qvbf, vbias, nullptr, nullptr);
    gemm_bf16_wmma<<<gq, 256, 0, stream>>>(ybf, Wkt, B_ * T_, D_, D_,
                                           kbf, nullptr, nullptr, nullptr, nullptr, nullptr);
    gemm_bf16_wmma<<<gq, 256, 0, stream>>>(ybf, Wvt, B_ * T_, D_, D_,
                                           vbf, nullptr, nullptr, nullptr, nullptr, nullptr);
    dim3 gp((L_ + 63) / 64, D_ / 32);
    gemm_bf16_wmma<<<gp, 256, 0, stream>>>(rlbf, Wpt, L_, D_, D_,
                                           pbf, nullptr, nullptr, nullptr, nullptr, nullptr);
    // 5) transpose v -> [B,H,K,T] (async global->LDS path)
    transpose_v<<<dim3(T_ / 32, K_ / 32, B_ * H_), 256, 0, stream>>>(vbf, vTb);
    // 6) fused rel-pos flash attention (TDM-staged K/V tiles)
    attn_kernel<<<dim3(T_ / 64, H_, B_), 128, 0, stream>>>(qubf, qvbf, kbf, pbf, vTb, obf);
    // 7) output projection + residual (fp32 out)
    gemm_bf16_wmma<<<gq, 256, 0, stream>>>(obf, Wot, B_ * T_, D_, D_,
                                           nullptr, nullptr, nullptr, nullptr, out, x);
}
